// MaskedAttentionDecoder_64020782514615
// MI455X (gfx1250) — matrix operand order repaired and checked by reading, not compiled, not run on hardware
//
#include <hip/hip_runtime.h>

// ---------------------------------------------------------------------------
// Types for CDNA5 WMMA (gfx1250, wave32)
// ---------------------------------------------------------------------------
typedef __attribute__((ext_vector_type(16))) __bf16 bf16x16;
typedef __attribute__((ext_vector_type(8)))  __bf16 bf16x8;
typedef __attribute__((ext_vector_type(8)))  float  f32x8;

#define B_    2
#define H_    256
#define NH_   8
#define DH_   32
#define DFF_  1024
#define PED_  256
#define NQ_   100

// Native bf16 convert (v_cvt_pk_bf16_f32 on gfx1250)
__device__ __forceinline__ __bf16 f2bf(float f) { return (__bf16)f; }

__device__ __forceinline__ void cvt4(bf16x16& d, int base, float4 f) {
  d[base + 0] = f2bf(f.x);
  d[base + 1] = f2bf(f.y);
  d[base + 2] = f2bf(f.z);
  d[base + 3] = f2bf(f.w);
}

// ---------------------------------------------------------------------------
// CDNA5 async global->LDS copy (ASYNCcnt). Probe-confirmed signature:
//   void __builtin_amdgcn_global_load_async_to_lds_b128(
//       int4 addrspace(1)* src, int4 addrspace(3)* dst, imm offset, imm cpol)
// ---------------------------------------------------------------------------
#if defined(__has_builtin)
#if __has_builtin(__builtin_amdgcn_global_load_async_to_lds_b128) && \
    __has_builtin(__builtin_amdgcn_s_wait_asynccnt)
#define USE_ASYNC_LDS 1
#endif
#endif
#ifndef USE_ASYNC_LDS
#define USE_ASYNC_LDS 0
#endif

#if USE_ASYNC_LDS
typedef int v4i __attribute__((__vector_size__(4 * sizeof(int))));
typedef __attribute__((address_space(1))) v4i* as1_v4i;
typedef __attribute__((address_space(3))) v4i* as3_v4i;
__device__ __forceinline__ void async_copy16(const float* g, float* l) {
  __builtin_amdgcn_global_load_async_to_lds_b128(
      (as1_v4i)(void*)g, (as3_v4i)(void*)l, 0, 0);
}
#endif

// ---------------------------------------------------------------------------
// Generic WMMA GEMM: C[M,N] = act(op(A)[M,K] @ W[K,N] + bias) (+ residual)
//   amode 0: A row-major (lda = row stride)
//   amode 1: A[m,k] = kfm[k*lda + m] (+ pos[m*K + k] if pos != null)
//   biasmode 0: none; 1: per-column bias[n]; 2: per-row bias[m*biasld]
// One wave (32 threads) per 16x16 tile, K stepped by 32.
// ---------------------------------------------------------------------------
__global__ __launch_bounds__(32)
void k_gemm(const float* __restrict__ A, const float* __restrict__ W,
            const float* __restrict__ bias, const float* __restrict__ pos,
            const float* __restrict__ residual, float* __restrict__ C,
            int M, int N, int K, int lda, int ldc,
            long strideA, long strideW, long strideC, long strideBias,
            int amode, int biasmode, int biasld, int act) {
  const int ln = threadIdx.x;
  const int tm = blockIdx.x * 16;
  const int tn = blockIdx.y * 16;
  const int b  = blockIdx.z;

  const float* Ab = A + (long)b * strideA;
  const float* Wb = W + (long)b * strideW;
  float*       Cb = C + (long)b * strideC;
  const float* biasb = bias ? bias + (long)b * strideBias : nullptr;
  const float* resb  = residual ? residual + (long)b * strideC : nullptr;

  __shared__ __align__(16) float sA[16 * 32];    // [m][k]
  __shared__ __align__(16) float sBt[16 * 32];   // [n][k]  (transposed W tile)

  f32x8 acc = {};
  const int row   = ln & 15;       // A row / B col / C col
  const int half  = ln >> 4;
  const int koffA = half * 16;
  const int koffB = half * 16;

  for (int k0 = 0; k0 < K; k0 += 32) {
    // ---- stage A tile (16x32 floats)
    bool asyncA = false;
#if USE_ASYNC_LDS
    asyncA = (amode == 0) && (tm + 16 <= M) && (k0 + 32 <= K) && ((lda & 3) == 0);
    if (asyncA) {
      // 128 x b128 transfers, 4 per lane
      for (int i = ln; i < 128; i += 32) {
        int m = i >> 3, kk = (i & 7) << 2;
        async_copy16(Ab + (long)(tm + m) * lda + (k0 + kk), sA + m * 32 + kk);
      }
    }
#endif
    if (!asyncA) {
      for (int i = ln; i < 512; i += 32) {
        int m = i >> 5, kk = i & 31;
        int gm = tm + m, gk = k0 + kk;
        float v = 0.0f;
        if (gm < M && gk < K) {
          if (amode == 0) v = Ab[(long)gm * lda + gk];
          else {
            v = Ab[(long)gk * lda + gm];
            if (pos) v += pos[(long)gm * K + gk];
          }
        }
        sA[m * 32 + kk] = v;
      }
    }
    // ---- stage B tile transposed: sBt[n][k]
    for (int i = ln; i < 512; i += 32) {
      int kk = i >> 4, n = i & 15;
      int gk = k0 + kk, gn = tn + n;
      float v = 0.0f;
      if (gk < K && gn < N) v = Wb[(long)gk * N + gn];
      sBt[n * 32 + kk] = v;
    }
    if (k0 + 32 < K)
      __builtin_prefetch(Wb + (long)(k0 + 32) * N + tn, 0, 1);
#if USE_ASYNC_LDS
    if (asyncA) __builtin_amdgcn_s_wait_asynccnt(0);
#endif
    __syncthreads();

    // ---- pack contiguous k = 16*half + s for A and B (vectorized LDS reads) and issue WMMA
    bf16x16 av, bv;
    {
      const float4* pa = (const float4*)(sA + row * 32 + koffA);
      float4 a0 = pa[0], a1 = pa[1];
      const float4* pa2 = (const float4*)(sA + row * 32 + koffA + 8);
      float4 a2 = pa2[0], a3 = pa2[1];
      cvt4(av, 0, a0); cvt4(av, 4, a1); cvt4(av, 8, a2); cvt4(av, 12, a3);
      const float4* pb = (const float4*)(sBt + row * 32 + koffB);
      float4 b0 = pb[0], b1 = pb[1], b2 = pb[2], b3 = pb[3];
      cvt4(bv, 0, b0); cvt4(bv, 4, b1); cvt4(bv, 8, b2); cvt4(bv, 12, b3);
    }
    acc = __builtin_amdgcn_wmma_f32_16x16x32_bf16(
        false, av, false, bv, (short)0, acc, false, false);
    __syncthreads();
  }

  // C/D layout: comp g -> M = g + 8*half, N = ln&15
#pragma unroll
  for (int g = 0; g < 8; ++g) {
    int gm = tm + g + half * 8;
    int gn = tn + row;
    if (gm < M && gn < N) {
      float v = acc[g];
      if (biasmode == 1)      v += biasb[gn];
      else if (biasmode == 2) v += biasb[(long)gm * biasld];
      if (act) v = fmaxf(v, 0.0f);
      if (resb) v += resb[(long)gm * ldc + gn];
      Cb[(long)gm * ldc + gn] = v;
    }
  }
}

// ---------------------------------------------------------------------------
// Fused multi-head attention (flash-style online softmax).
// q: (B,Lq,256)  k,v: (B,L,256)  bias: (B,Lq,L) of {0,-inf} or null
// out: (B,Lq,256).  grid = (ceil(Lq/16), NH, B), one wave per block.
// ---------------------------------------------------------------------------
__global__ __launch_bounds__(32)
void k_attn(const float* __restrict__ q, const float* __restrict__ k,
            const float* __restrict__ v, const float* __restrict__ bias,
            float* __restrict__ out, int L, int Lq) {
  const int ln    = threadIdx.x;
  const int m0    = blockIdx.x * 16;
  const int h     = blockIdx.y;
  const int b     = blockIdx.z;
  const int row   = ln & 15;
  const int half  = ln >> 4;
  const int koffA = half * 16;
  const int koffB = half * 16;
  const int nn    = ln & 15;

  const float* qp = q + ((long)b * Lq) * H_ + h * DH_;
  const float* kp = k + ((long)b * L) * H_ + h * DH_;
  const float* vp = v + ((long)b * L) * H_ + h * DH_;
  const float* bp = bias ? bias + (long)b * Lq * (long)L : nullptr;

  // Q tile in A layout (16x32 = full head dim), vectorized loads
  bf16x16 qa;
  {
    int gm = m0 + row;
    float4 z = {0.0f, 0.0f, 0.0f, 0.0f};
    float4 q0 = z, q1 = z, q2 = z, q3 = z;
    if (gm < Lq) {
      const float4* p = (const float4*)(qp + (long)gm * H_ + koffA);
      q0 = p[0]; q1 = p[1];
      const float4* p2 = (const float4*)(qp + (long)gm * H_ + koffA + 8);
      q2 = p2[0]; q3 = p2[1];
    }
    cvt4(qa, 0, q0); cvt4(qa, 4, q1); cvt4(qa, 8, q2); cvt4(qa, 12, q3);
  }

  f32x8 o0 = {}, o1 = {};
  float mstat[8], lstat[8];
#pragma unroll
  for (int g = 0; g < 8; ++g) { mstat[g] = -__builtin_inff(); lstat[g] = 0.0f; }

  __shared__ __align__(16) __bf16 sP[16][32];
  const float scale = 0.17677669529663687f;   // 1/sqrt(32)

  for (int kb0 = 0; kb0 < L; kb0 += 32) {
    // K tiles as B operand: B[kk][n] = K[kb0+n(+16)][kk]; 16 consecutive floats
    bf16x16 kt0, kt1;
    {
      int key0 = kb0 + nn;
      int key1 = kb0 + 16 + nn;
      float4 z = {0.0f, 0.0f, 0.0f, 0.0f};
      float4 a0 = z, a1 = z, a2 = z, a3 = z, c0 = z, c1 = z, c2 = z, c3 = z;
      if (key0 < L) {
        const float4* p = (const float4*)(kp + (long)key0 * H_ + koffB);
        a0 = p[0]; a1 = p[1]; a2 = p[2]; a3 = p[3];
      }
      if (key1 < L) {
        const float4* p = (const float4*)(kp + (long)key1 * H_ + koffB);
        c0 = p[0]; c1 = p[1]; c2 = p[2]; c3 = p[3];
      }
      cvt4(kt0, 0, a0); cvt4(kt0, 4, a1); cvt4(kt0, 8, a2); cvt4(kt0, 12, a3);
      cvt4(kt1, 0, c0); cvt4(kt1, 4, c1); cvt4(kt1, 8, c2); cvt4(kt1, 12, c3);
    }
    f32x8 zero = {};
    f32x8 s0 = __builtin_amdgcn_wmma_f32_16x16x32_bf16(
        false, qa, false, kt0, (short)0, zero, false, false);
    f32x8 s1 = __builtin_amdgcn_wmma_f32_16x16x32_bf16(
        false, qa, false, kt1, (short)0, zero, false, false);

    // scale + bias + padding mask + online softmax (row lives across 16 lanes)
#pragma unroll
    for (int g = 0; g < 8; ++g) {
      int r    = g + half * 8;
      int grow = m0 + r;
      int col0 = kb0 + nn;
      int col1 = kb0 + 16 + nn;
      float a0 = s0[g] * scale, a1 = s1[g] * scale;
      if (bp && grow < Lq) {
        if (col0 < L) a0 += bp[(long)grow * L + col0];
        if (col1 < L) a1 += bp[(long)grow * L + col1];
      }
      if (col0 >= L) a0 = -__builtin_inff();
      if (col1 >= L) a1 = -__builtin_inff();

      float mx = fmaxf(a0, a1);
#pragma unroll
      for (int s = 1; s <= 8; s <<= 1) mx = fmaxf(mx, __shfl_xor(mx, s, 32));
      float mold = mstat[g];
      float mnew = fmaxf(mold, mx);
      float alpha, p0, p1;
      if (mnew < -1e37f) { alpha = 1.0f; p0 = 0.0f; p1 = 0.0f; }
      else {
        alpha = (mold < -1e37f) ? 0.0f : __expf(mold - mnew);
        p0 = (a0 < -1e37f) ? 0.0f : __expf(a0 - mnew);
        p1 = (a1 < -1e37f) ? 0.0f : __expf(a1 - mnew);
      }
      float psum = p0 + p1;
#pragma unroll
      for (int s = 1; s <= 8; s <<= 1) psum += __shfl_xor(psum, s, 32);
      lstat[g] = lstat[g] * alpha + psum;
      mstat[g] = mnew;
      o0[g] *= alpha;
      o1[g] *= alpha;
      sP[r][nn]      = f2bf(p0);
      sP[r][nn + 16] = f2bf(p1);
    }
    __syncthreads();

    // reload P in A layout (vectorized); V tiles as B operand (strided gather)
    bf16x16 pa, vt0, vt1;
    {
      bf16x8 plo = *(const bf16x8*)(&sP[row][koffA]);
      bf16x8 phi = *(const bf16x8*)(&sP[row][koffA + 8]);
#pragma unroll
      for (int c = 0; c < 8; ++c) { pa[c] = plo[c]; pa[c + 8] = phi[c]; }
    }
#pragma unroll
    for (int c = 0; c < 16; ++c) {
      int key = kb0 + koffB + c;
      vt0[c] = f2bf((key < L) ? vp[(long)key * H_ + nn] : 0.0f);
      vt1[c] = f2bf((key < L) ? vp[(long)key * H_ + nn + 16] : 0.0f);
    }
    o0 = __builtin_amdgcn_wmma_f32_16x16x32_bf16(
        false, pa, false, vt0, (short)0, o0, false, false);
    o1 = __builtin_amdgcn_wmma_f32_16x16x32_bf16(
        false, pa, false, vt1, (short)0, o1, false, false);
    __syncthreads();
  }

  float* op = out + ((long)b * Lq) * H_ + h * DH_;
#pragma unroll
  for (int g = 0; g < 8; ++g) {
    int grow = m0 + g + half * 8;
    if (grow < Lq) {
      float inv = (lstat[g] > 0.0f) ? 1.0f / lstat[g] : 0.0f;
      op[(long)grow * H_ + nn]      = o0[g] * inv;
      op[(long)grow * H_ + nn + 16] = o1[g] * inv;
    }
  }
}

// ---------------------------------------------------------------------------
// LayerNorm over last dim (256). One 256-thread block per row.
// ---------------------------------------------------------------------------
__global__ __launch_bounds__(256)
void k_ln(const float* __restrict__ x, const float* __restrict__ g,
          const float* __restrict__ bt, float* __restrict__ y) {
  long row = blockIdx.x;
  int t = threadIdx.x;
  float v = x[row * H_ + t];
  __shared__ float red[256];
  red[t] = v; __syncthreads();
  for (int s = 128; s > 0; s >>= 1) { if (t < s) red[t] += red[t + s]; __syncthreads(); }
  float mean = red[0] / 256.0f;
  __syncthreads();
  float d = v - mean;
  red[t] = d * d; __syncthreads();
  for (int s = 128; s > 0; s >>= 1) { if (t < s) red[t] += red[t + s]; __syncthreads(); }
  float var = red[0] / 256.0f;
  y[row * H_ + t] = d * rsqrtf(var + 1e-5f) * g[t] + bt[t];
}

// ---------------------------------------------------------------------------
// Sine positional embedding (2D) + level embedding. pos: (h*w, 256)
// ---------------------------------------------------------------------------
__global__ __launch_bounds__(256)
void k_pos_embed(float* __restrict__ pos, const float* __restrict__ lvl_emb,
                 int h, int w) {
  int i = blockIdx.x * 256 + threadIdx.x;
  int total = h * w * H_;
  if (i >= total) return;
  int ch = i & 255;
  int pix = i >> 8;
  int r = pix / w, cl = pix % w;
  int cc = ch & 127;
  const float TWO_PI = 6.283185307179586f;
  float coord = (ch < 128) ? ((r + 1.0f) / h * TWO_PI) : ((cl + 1.0f) / w * TWO_PI);
  float dim_t = __powf(10000.0f, (float)(cc & ~1) / 128.0f);
  float p = coord / dim_t;
  float val = (cc & 1) ? __cosf(p) : __sinf(p);
  pos[i] = val + lvl_emb[ch];
}

// ---------------------------------------------------------------------------
// hs init: broadcast query embedding (100,256) -> (2,100,256)
// ---------------------------------------------------------------------------
__global__ __launch_bounds__(256)
void k_init_hs(const float* __restrict__ qe, float* __restrict__ hs) {
  int i = blockIdx.x * 256 + threadIdx.x;
  if (i < B_ * NQ_ * H_) hs[i] = qe[i % (NQ_ * H_)];
}

// ---------------------------------------------------------------------------
// Attention bias from mask logits: bilinear resize 128x128 -> (h,w),
// ignore where sigmoid(prob)<0.5 i.e. logit<0; all-ignored row -> no mask.
// grid = (NQ, B), 256 threads.
// ---------------------------------------------------------------------------
__global__ __launch_bounds__(256)
void k_maskbias(const float* __restrict__ ml, float* __restrict__ bias,
                int h, int w) {
  int q = blockIdx.x, b = blockIdx.y;
  const float* img = ml + ((long)b * NQ_ + q) * 16384;
  int L = h * w;
  float* bout = bias + ((long)b * NQ_ + q) * (long)L;
  __shared__ int s_cnt;
  if (threadIdx.x == 0) s_cnt = 0;
  __syncthreads();
  float sr_scale = 128.0f / h, sc_scale = 128.0f / w;
  int local = 0;
  for (int i = threadIdx.x; i < L; i += 256) {
    int r = i / w, c = i % w;
    float sr = fminf(fmaxf((r + 0.5f) * sr_scale - 0.5f, 0.0f), 127.0f);
    float sc = fminf(fmaxf((c + 0.5f) * sc_scale - 0.5f, 0.0f), 127.0f);
    int r0 = (int)floorf(sr); int r1 = min(r0 + 1, 127); float wr = sr - r0;
    int c0 = (int)floorf(sc); int c1 = min(c0 + 1, 127); float wc = sc - c0;
    float v0 = img[r0 * 128 + c0] * (1.0f - wr) + img[r1 * 128 + c0] * wr;
    float v1 = img[r0 * 128 + c1] * (1.0f - wr) + img[r1 * 128 + c1] * wr;
    float logit = v0 * (1.0f - wc) + v1 * wc;
    int ign = logit < 0.0f;
    bout[i] = ign ? -__builtin_inff() : 0.0f;
    local += ign;
  }
  atomicAdd(&s_cnt, local);
  __syncthreads();
  if (s_cnt == L) {
    for (int i = threadIdx.x; i < L; i += 256) bout[i] = 0.0f;
  }
}

// ---------------------------------------------------------------------------
// Host orchestration
// ---------------------------------------------------------------------------
extern "C" void kernel_launch(void* const* d_in, const int* in_sizes, int n_in,
                              void* d_out, int out_size, void* d_ws, size_t ws_size,
                              hipStream_t stream) {
  (void)in_sizes; (void)n_in; (void)out_size; (void)ws_size;
  auto P = [&](int i) { return (const float*)d_in[i]; };

  const int Lh[3] = {32, 64, 128};
  const int Lw[3] = {32, 64, 128};
  const int Ls[3] = {1024, 4096, 16384};
  const int LB = 4, PER_LAYER = 26;
  const int IDX_LEVEL_EMB = 238;   // (3,256)
  const int MP = 239;              // W1 W2 W3 b1 b2 b3 ln_b ln_g
  const int IDX_QUERY = 247;       // (100,256)

  size_t off = 0;
  auto alloc = [&](size_t nf) {
    float* p = (float*)((char*)d_ws + off);
    off += ((nf * sizeof(float)) + 255) / 256 * 256;
    return p;
  };
  float* pos[3];
  for (int l = 0; l < 3; ++l) pos[l] = alloc((size_t)Ls[l] * H_);
  float* hs   = alloc(B_ * NQ_ * H_);
  float* xb   = alloc(B_ * NQ_ * H_);
  float* qb   = alloc(B_ * NQ_ * H_);
  float* kbuf = alloc((size_t)B_ * 16384 * H_);
  float* vbuf = alloc((size_t)B_ * 16384 * H_);
  float* ksb  = alloc(B_ * NQ_ * H_);
  float* vsb  = alloc(B_ * NQ_ * H_);
  float* att  = alloc(B_ * NQ_ * H_);
  float* hid  = alloc(B_ * NQ_ * DFF_);
  float* wbuf = alloc(B_ * NQ_ * 257);
  float* h1   = alloc(B_ * NQ_ * H_);
  float* h2   = alloc(B_ * NQ_ * H_);
  float* mb   = alloc((size_t)B_ * NQ_ * 16384);

  float* embeds_out = (float*)d_out;
  float* logits_out = embeds_out + (size_t)10 * B_ * NQ_ * H_;
  const float* pe = P(3);

  auto gemm = [&](const float* A, const float* W, const float* bias,
                  const float* posp, const float* res, float* C,
                  int M, int N, int K, int lda, int ldc,
                  long sA, long sW, long sC, long sBias,
                  int amode, int biasmode, int biasld, int act) {
    dim3 g((M + 15) / 16, (N + 15) / 16, B_);
    k_gemm<<<g, 32, 0, stream>>>(A, W, bias, posp, res, C, M, N, K, lda, ldc,
                                 sA, sW, sC, sBias, amode, biasmode, biasld, act);
  };

  auto mask_pred = [&](int slot) {
    k_ln<<<B_ * NQ_, 256, 0, stream>>>(hs, P(MP + 7), P(MP + 6), xb);
    gemm(xb, P(MP + 0), P(MP + 3), nullptr, nullptr, h1,
         NQ_, H_, H_, H_, H_, NQ_ * H_, 0, NQ_ * H_, 0, 0, 1, 0, 1);
    gemm(h1, P(MP + 1), P(MP + 4), nullptr, nullptr, h2,
         NQ_, H_, H_, H_, H_, NQ_ * H_, 0, NQ_ * H_, 0, 0, 1, 0, 1);
    gemm(h2, P(MP + 2), P(MP + 5), nullptr, nullptr, wbuf,
         NQ_, 257, H_, H_, 257, NQ_ * H_, 0, NQ_ * 257, 0, 0, 1, 0, 0);
    float* ml = logits_out + (size_t)slot * B_ * NQ_ * 16384;
    gemm(wbuf, pe, wbuf + PED_, nullptr, nullptr, ml,
         NQ_, 16384, PED_, 257, 16384,
         (long)NQ_ * 257, (long)PED_ * 16384, (long)NQ_ * 16384, (long)NQ_ * 257,
         0, 2, 257, 0);
  };

  // positional embeddings (+ level embedding folded in)
  for (int l = 0; l < 3; ++l) {
    int total = Ls[l] * H_;
    k_pos_embed<<<(total + 255) / 256, 256, 0, stream>>>(
        pos[l], P(IDX_LEVEL_EMB) + l * H_, Lh[l], Lw[l]);
  }
  k_init_hs<<<(B_ * NQ_ * H_ + 255) / 256, 256, 0, stream>>>(P(IDX_QUERY), hs);
  (void)hipMemcpyAsync(embeds_out, hs, (size_t)B_ * NQ_ * H_ * sizeof(float),
                       hipMemcpyDeviceToDevice, stream);
  mask_pred(0);

  for (int idx = 0; idx < 9; ++idx) {
    int lvl = idx % 3;
    int L = Ls[lvl];
    const float* kfm = P(lvl);
    int lb = LB + idx * PER_LAYER;
    bool masked = idx >= 3;

    if (masked) {
      const float* ml = logits_out + (size_t)idx * B_ * NQ_ * 16384;
      dim3 g(NQ_, B_);
      k_maskbias<<<g, 256, 0, stream>>>(ml, mb, Lh[lvl], Lw[lvl]);
    }

    // ---- cross attention: x=LN(hs); hs += MHA(x, key_hs+pos, key_hs) @ Wo
    k_ln<<<B_ * NQ_, 256, 0, stream>>>(hs, P(lb + 9), P(lb + 8), xb);
    gemm(xb, P(lb + 2), P(lb + 6), nullptr, nullptr, qb,
         NQ_, H_, H_, H_, H_, NQ_ * H_, 0, NQ_ * H_, 0, 0, 1, 0, 0);
    gemm(kfm, P(lb + 0), P(lb + 4), pos[lvl], nullptr, kbuf,
         L, H_, H_, L, H_, (long)H_ * L, 0, (long)L * H_, 0, 1, 1, 0, 0);
    gemm(kfm, P(lb + 3), P(lb + 7), nullptr, nullptr, vbuf,
         L, H_, H_, L, H_, (long)H_ * L, 0, (long)L * H_, 0, 1, 1, 0, 0);
    {
      dim3 g((NQ_ + 15) / 16, NH_, B_);
      k_attn<<<g, 32, 0, stream>>>(qb, kbuf, vbuf, masked ? mb : nullptr,
                                   att, L, NQ_);
    }
    gemm(att, P(lb + 1), P(lb + 5), nullptr, hs, hs,
         NQ_, H_, H_, H_, H_, NQ_ * H_, 0, NQ_ * H_, 0, 0, 1, 0, 0);

    // ---- self attention
    k_ln<<<B_ * NQ_, 256, 0, stream>>>(hs, P(lb + 25), P(lb + 24), xb);
    gemm(xb, P(lb + 18), P(lb + 22), nullptr, nullptr, qb,
         NQ_, H_, H_, H_, H_, NQ_ * H_, 0, NQ_ * H_, 0, 0, 1, 0, 0);
    gemm(xb, P(lb + 16), P(lb + 20), nullptr, nullptr, ksb,
         NQ_, H_, H_, H_, H_, NQ_ * H_, 0, NQ_ * H_, 0, 0, 1, 0, 0);
    gemm(xb, P(lb + 19), P(lb + 23), nullptr, nullptr, vsb,
         NQ_, H_, H_, H_, H_, NQ_ * H_, 0, NQ_ * H_, 0, 0, 1, 0, 0);
    {
      dim3 g((NQ_ + 15) / 16, NH_, B_);
      k_attn<<<g, 32, 0, stream>>>(qb, ksb, vsb, nullptr, att, NQ_, NQ_);
    }
    gemm(att, P(lb + 17), P(lb + 21), nullptr, hs, hs,
         NQ_, H_, H_, H_, H_, NQ_ * H_, 0, NQ_ * H_, 0, 0, 1, 0, 0);

    // ---- MLP
    k_ln<<<B_ * NQ_, 256, 0, stream>>>(hs, P(lb + 11), P(lb + 10), xb);
    gemm(xb, P(lb + 12), P(lb + 14), nullptr, nullptr, hid,
         NQ_, DFF_, H_, H_, DFF_, NQ_ * H_, 0, NQ_ * DFF_, 0, 0, 1, 0, 1);
    gemm(hid, P(lb + 13), P(lb + 15), nullptr, hs, hs,
         NQ_, H_, DFF_, DFF_, H_, NQ_ * DFF_, 0, NQ_ * H_, 0, 0, 1, 0, 0);

    (void)hipMemcpyAsync(embeds_out + (size_t)(idx + 1) * B_ * NQ_ * H_, hs,
                         (size_t)B_ * NQ_ * H_ * sizeof(float),
                         hipMemcpyDeviceToDevice, stream);
    mask_pred(idx + 1);
  }
}
